// MyImprovedClusteredAttention_13211319403254
// MI455X (gfx1250) — matrix operand order repaired and checked by reading, not compiled, not run on hardware
//
#include <hip/hip_runtime.h>
#include <hip/hip_bf16.h>
#include <math.h>

// ---- problem constants (match reference) ------------------------------------
constexpr int N_ = 2, H_ = 8, L_ = 2048, S_ = 2048, E_ = 64, D_ = 64;
constexpr int C_ = 128, K_ = 32, BITS_ = 32, ITERS_ = 10;
constexpr int NH_ = N_ * H_;
constexpr float TEMP_ = 0.125f;  // 1/sqrt(E)

typedef __attribute__((ext_vector_type(16))) _Float16 v16h;
typedef __attribute__((ext_vector_type(8)))  float    v8f;

// =============================================================================
// K1: hash projections -> packed 32-bit codes per (n,h,l)
// =============================================================================
__global__ void hash_kernel(const float* __restrict__ q,
                            const float* __restrict__ planes,
                            unsigned* __restrict__ bits) {
  int t = blockIdx.x * blockDim.x + threadIdx.x;
  if (t >= NH_ * L_) return;
  int nh = t / L_, l = t % L_;
  int n = nh / H_, h = nh % H_;
  const float* qp = q + ((size_t)(n * L_ + l) * H_ + h) * E_;
  float qv[E_];
#pragma unroll
  for (int e = 0; e < E_; ++e) qv[e] = qp[e];
  unsigned w = 0;
  for (int b = 0; b < BITS_; ++b) {
    const float* pl = planes + b * (E_ + 1);
    float acc = pl[E_];
#pragma unroll
    for (int e = 0; e < E_; ++e) acc += qv[e] * pl[e];
    if (acc > 0.f) w |= (1u << b);
  }
  bits[t] = w;
}

// =============================================================================
// K2: centroid init: cbits0 = bits at linspace(0, L-1, C).astype(int32)
// =============================================================================
__global__ void init_cent_kernel(const unsigned* __restrict__ bits,
                                 unsigned* __restrict__ cbits) {
  int t = blockIdx.x * blockDim.x + threadIdx.x;
  if (t >= NH_ * C_) return;
  int nh = t / C_, c = t % C_;
  int idx = (int)((double)c * (double)(L_ - 1) / (double)(C_ - 1));
  cbits[t] = bits[nh * L_ + idx];
}

// =============================================================================
// K3: one Lloyd iteration over Hamming distance (popcount on packed bits).
//     d_float = popc(q)+popc(c)-2*popc(q&c) == popc(q^c) exactly, so argmin
//     with strict '<' over ascending c matches jnp.argmin tie-breaking.
// =============================================================================
__global__ void kmeans_iter_kernel(const unsigned* __restrict__ bits,
                                   unsigned* __restrict__ cbits) {
  int nh = blockIdx.x;
  __shared__ unsigned cent[C_];
  __shared__ unsigned sums[C_][BITS_];
  __shared__ unsigned cnt[C_];
  int tid = threadIdx.x;
  for (int c = tid; c < C_; c += 256) { cent[c] = cbits[nh * C_ + c]; cnt[c] = 0; }
  for (int i = tid; i < C_ * BITS_; i += 256) (&sums[0][0])[i] = 0;
  __syncthreads();
  for (int l = tid; l < L_; l += 256) {
    unsigned w = bits[nh * L_ + l];
    int best = 0, bd = 1 << 30;
    for (int c = 0; c < C_; ++c) {
      int d = __popc(w ^ cent[c]);
      if (d < bd) { bd = d; best = c; }
    }
    atomicAdd(&cnt[best], 1u);
    for (int b = 0; b < BITS_; ++b)
      if ((w >> b) & 1u) atomicAdd(&sums[best][b], 1u);
  }
  __syncthreads();
  for (int c = tid; c < C_; c += 256) {
    unsigned n = cnt[c];
    unsigned nc = cent[c];
    if (n > 0) {
      nc = 0;
      for (int b = 0; b < BITS_; ++b)
        if (2u * sums[c][b] > n) nc |= (1u << b);
    }
    cbits[nh * C_ + c] = nc;
  }
}

// =============================================================================
// K4: final assignment + cluster means Qg = sum(q)/max(count,1) via LDS f32 adds
// =============================================================================
__global__ void final_assign_kernel(const unsigned* __restrict__ bits,
                                    const unsigned* __restrict__ cbits,
                                    const float* __restrict__ q,
                                    int* __restrict__ cl, int* __restrict__ cnts,
                                    float* __restrict__ Qg) {
  int nh = blockIdx.x;
  int n = nh / H_, h = nh % H_;
  __shared__ unsigned cent[C_];
  __shared__ float qsum[C_ * E_];
  __shared__ int cnt[C_];
  int tid = threadIdx.x;
  for (int c = tid; c < C_; c += 256) { cent[c] = cbits[nh * C_ + c]; cnt[c] = 0; }
  for (int i = tid; i < C_ * E_; i += 256) qsum[i] = 0.f;
  __syncthreads();
  for (int l = tid; l < L_; l += 256) {
    unsigned w = bits[nh * L_ + l];
    int best = 0, bd = 1 << 30;
    for (int c = 0; c < C_; ++c) {
      int d = __popc(w ^ cent[c]);
      if (d < bd) { bd = d; best = c; }
    }
    cl[nh * L_ + l] = best;
    atomicAdd(&cnt[best], 1);
    const float* qp = q + ((size_t)(n * L_ + l) * H_ + h) * E_;
#pragma unroll
    for (int e = 0; e < E_; ++e) atomicAdd(&qsum[best * E_ + e], qp[e]);
  }
  __syncthreads();
  for (int i = tid; i < C_ * E_; i += 256) {
    int c = i / E_;
    float cs = (float)cnt[c]; if (cs < 1.f) cs = 1.f;
    Qg[(size_t)nh * C_ * E_ + i] = qsum[i] / cs;
  }
  for (int c = tid; c < C_; c += 256) cnts[nh * C_ + c] = cnt[c];
}

// =============================================================================
// K5: stable counting sort by cluster id (matches stable jnp.argsort).
// =============================================================================
__global__ void sort_kernel(const int* __restrict__ cl, const int* __restrict__ cnts,
                            int* __restrict__ sidx, int* __restrict__ scl) {
  if (threadIdx.x != 0) return;
  int nh = blockIdx.x;
  int off[C_];
  int acc = 0;
  for (int c = 0; c < C_; ++c) { off[c] = acc; acc += cnts[nh * C_ + c]; }
  for (int l = 0; l < L_; ++l) {
    int c = cl[nh * L_ + l];
    int p = off[c]++;
    sidx[nh * L_ + p] = l;
    scl[nh * L_ + p] = c;
  }
}

// =============================================================================
// K6: QK[nh] = Qg(128x64) * K^T(64x2048) via v_wmma_f32_16x16x32_f16.
// =============================================================================
__global__ void qk_wmma_kernel(const float* __restrict__ Qg,
                               const float* __restrict__ keys,
                               float* __restrict__ QK) {
  int bx = blockIdx.x;
  int nh = bx >> 3;          // 8 c-tiles per head
  int ct = bx & 7;
  int n = nh / H_, h = nh % H_;
  int c0 = ct * 16;
  int tid = threadIdx.x;
  int lane = tid & 31;
  int wid = tid >> 5;
  int mrow = lane & 15;
  int half = lane >> 4;

  // A fragments (16x32 f16): lane row = c0+mrow; half selects K sub-ranges.
  const float* qgp = Qg + ((size_t)nh * C_ + c0 + mrow) * E_;
  v16h a0, a1;
#pragma unroll
  for (int j = 0; j < 8; ++j) {
    a0[j]     = (_Float16)qgp[      half * 16 + j];
    a0[8 + j] = (_Float16)qgp[8   + half * 16 + j];
    a1[j]     = (_Float16)qgp[32  + half * 16 + j];
    a1[8 + j] = (_Float16)qgp[40  + half * 16 + j];
  }

  float* qkbase = QK + (size_t)nh * C_ * S_;
  for (int st = wid; st < S_ / 16; st += 8) {
    int sb = st * 16;
    // B fragment (32x16 f16): column = mrow -> 16 contiguous f32 from key row.
    const float* kp = keys + ((size_t)(n * S_ + sb + mrow) * H_ + h) * E_;
    if (st + 8 < S_ / 16)  // global_prefetch_b8 for the next S-tile
      __builtin_prefetch(keys + ((size_t)(n * S_ + sb + 128 + mrow) * H_ + h) * E_, 0, 1);
    v16h b0, b1;
#pragma unroll
    for (int j = 0; j < 16; ++j) {
      b0[j] = (_Float16)kp[     half * 16 + j];
      b1[j] = (_Float16)kp[32 + half * 16 + j];
    }
    v8f acc = {};
    acc = __builtin_amdgcn_wmma_f32_16x16x32_f16(false, a0, false, b0, (short)0, acc, false, false);
    acc = __builtin_amdgcn_wmma_f32_16x16x32_f16(false, a1, false, b1, (short)0, acc, false, false);
#pragma unroll
    for (int i = 0; i < 8; ++i)
      qkbase[(size_t)(c0 + i + half * 8) * S_ + sb + mrow] = acc[i];
  }
}

// =============================================================================
// K7: per (n,h,c): top-32 (first-index ties like lax.top_k), full softmax,
//     bottom-k mass.  A_b row is written back IN PLACE into the QK scratch so
//     the bottom-k aggregation can run as a WMMA GEMM (K7b) instead of each
//     cluster re-reading the whole V matrix (~16x less L2 traffic).
// =============================================================================
__global__ void topk_kernel(float* __restrict__ QK,
                            float* __restrict__ tkvg, int* __restrict__ tkig,
                            float* __restrict__ abkg) {
  int nhc = blockIdx.x;
  int tid = threadIdx.x;
  __shared__ float work[S_];
  __shared__ unsigned char flag[S_];
  __shared__ float redv[256];
  __shared__ int   redi[256];
  __shared__ float red2[256];
  __shared__ float tkv[K_];
  __shared__ int   tki[K_];
  __shared__ float s_sum;

  float* row = QK + (size_t)nhc * S_;
  for (int s = tid; s < S_; s += 256) { work[s] = row[s]; flag[s] = 0; }
  __syncthreads();

  for (int kk = 0; kk < K_; ++kk) {
    float bv = -INFINITY; int bi = S_;
    for (int s = tid; s < S_; s += 256) {
      float v = work[s];
      if (v > bv || (v == bv && s < bi)) { bv = v; bi = s; }
    }
    redv[tid] = bv; redi[tid] = bi;
    __syncthreads();
    for (int st = 128; st > 0; st >>= 1) {
      if (tid < st) {
        float ov = redv[tid + st]; int oi = redi[tid + st];
        if (ov > redv[tid] || (ov == redv[tid] && oi < redi[tid])) { redv[tid] = ov; redi[tid] = oi; }
      }
      __syncthreads();
    }
    if (tid == 0) {
      tkv[kk] = redv[0]; tki[kk] = redi[0];
      flag[redi[0]] = 1; work[redi[0]] = -INFINITY;
    }
    __syncthreads();
  }

  if (tid < K_) work[tki[tid]] = tkv[tid];   // restore raw values
  __syncthreads();

  float m = TEMP_ * tkv[0];                  // global max (TEMP > 0)
  float part = 0.f;
  for (int s = tid; s < S_; s += 256) {
    float e = __expf(TEMP_ * work[s] - m);
    work[s] = e; part += e;
  }
  red2[tid] = part; __syncthreads();
  for (int st = 128; st > 0; st >>= 1) { if (tid < st) red2[tid] += red2[tid + st]; __syncthreads(); }
  if (tid == 0) s_sum = red2[0];
  __syncthreads();
  float inv = 1.f / s_sum;
  part = 0.f;
  for (int s = tid; s < S_; s += 256) {
    float ab = flag[s] ? 0.f : work[s] * inv;  // A_b = softmax * (1 - topk mask)
    row[s] = ab;                               // write A_b back to global scratch
    part += ab;
  }
  red2[tid] = part; __syncthreads();
  for (int st = 128; st > 0; st >>= 1) { if (tid < st) red2[tid] += red2[tid + st]; __syncthreads(); }
  if (tid == 0) abkg[nhc] = red2[0];
  if (tid < K_) {
    tkvg[(size_t)nhc * K_ + tid] = tkv[tid];
    tkig[(size_t)nhc * K_ + tid] = tki[tid];
  }
}

// =============================================================================
// K7b: Vb_g(128x64) = A_b(128x2048) * V(2048x64) per head, via WMMA.
//      Block = one (head, c-tile); 8 waves split K=2048 (8 K-iters each),
//      each wave holds 4 accumulator tiles (d-tiles of 16), partials reduced
//      through LDS ds_add_f32.
// =============================================================================
__global__ void vb_wmma_kernel(const float* __restrict__ Ab,   // QK scratch, now A_b
                               const float* __restrict__ values,
                               float* __restrict__ vbg) {
  int bx = blockIdx.x;
  int nh = bx >> 3;
  int ct = bx & 7;
  int n = nh / H_, h = nh % H_;
  int tid = threadIdx.x;
  int lane = tid & 31;
  int wid = tid >> 5;
  int mrow = lane & 15;
  int half = lane >> 4;

  __shared__ float red[16 * D_];
  for (int i = tid; i < 16 * D_; i += 256) red[i] = 0.f;
  __syncthreads();

  const float* arow = Ab + ((size_t)nh * C_ + ct * 16 + mrow) * S_;
  v8f acc0 = {}, acc1 = {}, acc2 = {}, acc3 = {};
  int sbeg = wid * (S_ / 8);
  for (int s0 = sbeg; s0 < sbeg + S_ / 8; s0 += 32) {
    // A fragment (16x32 f16) from A_b: contiguous along K.
    v16h a;
#pragma unroll
    for (int j = 0; j < 8; ++j) {
      a[j]     = (_Float16)arow[s0 +     half * 16 + j];
      a[8 + j] = (_Float16)arow[s0 + 8 + half * 16 + j];
    }
    // B fragments (32x16 f16) from V: column = d-tile*16 + mrow, K strided by H*D.
    const float* vrow = values + ((size_t)(n * S_ + s0 + half * 16) * H_ + h) * D_;
    v16h b0, b1, b2, b3;
#pragma unroll
    for (int j = 0; j < 16; ++j) {
      const float* vr = vrow + (size_t)j * H_ * D_;
      b0[j] = (_Float16)vr[ 0 + mrow];
      b1[j] = (_Float16)vr[16 + mrow];
      b2[j] = (_Float16)vr[32 + mrow];
      b3[j] = (_Float16)vr[48 + mrow];
    }
    acc0 = __builtin_amdgcn_wmma_f32_16x16x32_f16(false, a, false, b0, (short)0, acc0, false, false);
    acc1 = __builtin_amdgcn_wmma_f32_16x16x32_f16(false, a, false, b1, (short)0, acc1, false, false);
    acc2 = __builtin_amdgcn_wmma_f32_16x16x32_f16(false, a, false, b2, (short)0, acc2, false, false);
    acc3 = __builtin_amdgcn_wmma_f32_16x16x32_f16(false, a, false, b3, (short)0, acc3, false, false);
  }
  // Cross-wave reduction of partial tiles (C/D layout: row = i + 8*half, col = mrow)
#pragma unroll
  for (int i = 0; i < 8; ++i) {
    int r = i + half * 8;
    atomicAdd(&red[r * D_ +  0 + mrow], acc0[i]);
    atomicAdd(&red[r * D_ + 16 + mrow], acc1[i]);
    atomicAdd(&red[r * D_ + 32 + mrow], acc2[i]);
    atomicAdd(&red[r * D_ + 48 + mrow], acc3[i]);
  }
  __syncthreads();
  float* out = vbg + ((size_t)nh * C_ + ct * 16) * D_;
  for (int i = tid; i < 16 * D_; i += 256) out[i] = red[i];
}

// =============================================================================
// K8: one wave32 per sorted query: lane k owns selected key k.
// =============================================================================
__global__ void final_attn_kernel(const float* __restrict__ q,
                                  const float* __restrict__ keys,
                                  const float* __restrict__ values,
                                  const int* __restrict__ sidx,
                                  const int* __restrict__ scl,
                                  const float* __restrict__ topkv,
                                  const int* __restrict__ topki,
                                  const float* __restrict__ abk,
                                  const float* __restrict__ vbg,
                                  float* __restrict__ vout,
                                  int* __restrict__ siout,
                                  float* __restrict__ atout) {
  int tid = threadIdx.x;
  int lane = tid & 31;
  int wid = tid >> 5;
  int p = blockIdx.x * 8 + wid;          // sorted-position index over NH*L
  int i = p % L_;
  int nh = p / L_;
  int n = nh / H_, h = nh % H_;
  int c = scl[nh * L_ + i];
  int lorig = sidx[nh * L_ + i];
  int k = lane;
  int ki = topki[((size_t)nh * C_ + c) * K_ + k];
  float tv0 = topkv[(size_t)n * H_ * C_ * K_ + k];  // topk_values[n,0,0,k] inf mask

  const float* qp = q + ((size_t)(n * L_ + lorig) * H_ + h) * E_;
  const float* kp = keys + ((size_t)(n * S_ + ki) * H_ + h) * E_;
  float dot = 0.f;
#pragma unroll
  for (int e = 0; e < E_; ++e) dot += qp[e] * kp[e];
  float x = isinf(tv0) ? -INFINITY : TEMP_ * dot;

  float m = x;
  for (int off = 16; off > 0; off >>= 1) m = fmaxf(m, __shfl_xor(m, off, 32));
  float e = __expf(x - m);
  float s = e;
  for (int off = 16; off > 0; off >>= 1) s += __shfl_xor(s, off, 32);
  float a = (e / s) * (1.f - abk[nh * C_ + c]);

  const float* vp = values + ((size_t)(n * S_ + ki) * H_ + h) * D_;
  const float* vb = vbg + ((size_t)nh * C_ + c) * D_;
  float* vo = vout + ((size_t)(n * L_ + lorig) * H_ + h) * D_;
#pragma unroll 4
  for (int d = 0; d < D_; ++d) {
    float t = a * vp[d];
    for (int off = 16; off > 0; off >>= 1) t += __shfl_xor(t, off, 32);
    if (lane == 0) vo[d] = t + vb[d];
  }
  size_t ob = ((size_t)nh * L_ + lorig) * K_ + k;
  siout[ob] = ki;
  atout[ob] = a;
}

// =============================================================================
extern "C" void kernel_launch(void* const* d_in, const int* in_sizes, int n_in,
                              void* d_out, int out_size, void* d_ws, size_t ws_size,
                              hipStream_t stream) {
  const float* q      = (const float*)d_in[0];
  const float* keys   = (const float*)d_in[1];
  const float* values = (const float*)d_in[2];
  const float* planes = (const float*)d_in[3];

  char* ws = (char*)d_ws;
  size_t cur = 0;
  auto alloc = [&](size_t bytes) -> void* {
    size_t a = (cur + 255) & ~size_t(255);
    cur = a + bytes;
    return ws + a;
  };
  unsigned* bits  = (unsigned*)alloc((size_t)NH_ * L_ * 4);
  unsigned* cbits = (unsigned*)alloc((size_t)NH_ * C_ * 4);
  int*      cl    = (int*)     alloc((size_t)NH_ * L_ * 4);
  int*      cnts  = (int*)     alloc((size_t)NH_ * C_ * 4);
  float*    Qg    = (float*)   alloc((size_t)NH_ * C_ * E_ * 4);
  float*    QK    = (float*)   alloc((size_t)NH_ * C_ * S_ * 4);   // 16 MB; later holds A_b
  float*    tkv   = (float*)   alloc((size_t)NH_ * C_ * K_ * 4);
  int*      tki   = (int*)     alloc((size_t)NH_ * C_ * K_ * 4);
  float*    abk   = (float*)   alloc((size_t)NH_ * C_ * 4);
  float*    vbg   = (float*)   alloc((size_t)NH_ * C_ * D_ * 4);
  int*      sidx  = (int*)     alloc((size_t)NH_ * L_ * 4);
  int*      scl   = (int*)     alloc((size_t)NH_ * L_ * 4);

  float* outf  = (float*)d_out;
  float* vout  = outf;                                          // [N,L,H,D]
  int*   siout = (int*)(outf + (size_t)N_ * L_ * H_ * D_);      // [N,H,L,K] int32
  float* atout = outf + (size_t)N_ * L_ * H_ * D_ + (size_t)N_ * H_ * L_ * K_;  // [N,H,L,K]

  hash_kernel<<<(NH_ * L_ + 255) / 256, 256, 0, stream>>>(q, planes, bits);
  init_cent_kernel<<<(NH_ * C_ + 255) / 256, 256, 0, stream>>>(bits, cbits);
  for (int it = 0; it < ITERS_; ++it)
    kmeans_iter_kernel<<<NH_, 256, 0, stream>>>(bits, cbits);
  final_assign_kernel<<<NH_, 256, 0, stream>>>(bits, cbits, q, cl, cnts, Qg);
  sort_kernel<<<NH_, 32, 0, stream>>>(cl, cnts, sidx, scl);
  qk_wmma_kernel<<<NH_ * (C_ / 16), 256, 0, stream>>>(Qg, keys, QK);
  topk_kernel<<<NH_ * C_, 256, 0, stream>>>(QK, tkv, tki, abk);
  vb_wmma_kernel<<<NH_ * (C_ / 16), 256, 0, stream>>>(QK, values, vbg);
  final_attn_kernel<<<(NH_ * L_) / 8, 256, 0, stream>>>(
      q, keys, values, sidx, scl, tkv, tki, abk, vbg, vout, siout, atout);
}
